// AttentionGNN_19155554140460
// MI455X (gfx1250) — compile-verified, matrix operand order checked by reading, not run on hardware
//
#include <hip/hip_runtime.h>
#include <hip/hip_fp16.h>
#include <math.h>

typedef _Float16 half8  __attribute__((ext_vector_type(8)));
typedef _Float16 half16 __attribute__((ext_vector_type(16)));
typedef float    float8 __attribute__((ext_vector_type(8)));

#define N_NODES 50000
#define E_EDGES 800000
#define E_TOT   850000   // E + N self loops
#define IN_CH   128
#define D1      128      // HEADS*HID
#define HEADS1  4
#define HID1    32
#define OUT_CH  32
#define NEG_SLOPE 0.2f

// ---------------- utility kernels ----------------

__global__ void k_f32_to_f16(_Float16* __restrict__ dst, const float* __restrict__ src, int n) {
  int i = blockIdx.x * blockDim.x + threadIdx.x;
  if (i < n) dst[i] = (_Float16)src[i];
}

// src[R][C] (f32, row major) -> dst[C][R] (f16, transposed)
__global__ void k_f32_to_f16_T(_Float16* __restrict__ dst, const float* __restrict__ src,
                               int R, int Ccols) {
  int i = blockIdx.x * blockDim.x + threadIdx.x;
  if (i >= R * Ccols) return;
  int r = i / Ccols, c = i - r * Ccols;
  dst[(size_t)c * R + r] = (_Float16)src[(size_t)r * Ccols + c];
}

__global__ void k_fill(float* __restrict__ p, float v, int n) {
  int i = blockIdx.x * blockDim.x + threadIdx.x;
  if (i < n) p[i] = v;
}

__global__ void k_bias_fill(float* __restrict__ p, const float* __restrict__ b, int n, int C) {
  int i = blockIdx.x * blockDim.x + threadIdx.x;
  if (i < n) p[i] = b[i % C];
}

// ---------------- WMMA GEMM: C[M,Nc] = A[M,K] x Bt[Nc,K]^T (f16 in, f32 out) ------------
// Bt is the weight matrix pre-transposed to [Nc][K] so both A and B operands are two
// contiguous b128 loads per lane per 16x16x32 step.
// One wave computes a 16 x (16*CT) strip.  M%16==0, K%32==0, Nc%(16*CT)==0.
template <int K, int CT>
__global__ void k_gemm_wmma(float* __restrict__ C, const _Float16* __restrict__ A,
                            const _Float16* __restrict__ Bt, int M, int Nc) {
  int wave = threadIdx.x >> 5;
  int lane = threadIdx.x & 31;
  int colGroups = Nc / (16 * CT);
  int nTiles    = (M >> 4) * colGroups;
  int tile = blockIdx.x * (blockDim.x >> 5) + wave;
  if (tile >= nTiles) return;                 // wave-uniform guard: EXEC stays all-ones
  int rowT = tile / colGroups;
  int cg   = tile - rowT * colGroups;
  int row  = (rowT << 4) + (lane & 15);
  int koff = (lane >> 4) << 3;                // K-striping: 0 (lanes 0-15) / 8 (lanes 16-31)

  const _Float16* ap = A + (size_t)row * K + koff;
  const _Float16* bp[CT];
#pragma unroll
  for (int t = 0; t < CT; ++t) {
    int col = ((cg * CT + t) << 4) + (lane & 15);
    bp[t] = Bt + (size_t)col * K + koff;
  }

  float8 acc[CT];
#pragma unroll
  for (int t = 0; t < CT; ++t) acc[t] = (float8){};

#pragma unroll
  for (int kk = 0; kk < K; kk += 32) {
    half8 alo = *(const half8*)(ap + kk);
    half8 ahi = *(const half8*)(ap + kk + 16);
    half16 a;
#pragma unroll
    for (int i = 0; i < 8; ++i) { a[i] = alo[i]; a[i + 8] = ahi[i]; }
#pragma unroll
    for (int t = 0; t < CT; ++t) {
      half8 blo = *(const half8*)(bp[t] + kk);
      half8 bhi = *(const half8*)(bp[t] + kk + 16);
      half16 b;
#pragma unroll
      for (int i = 0; i < 8; ++i) { b[i] = blo[i]; b[i + 8] = bhi[i]; }
      acc[t] = __builtin_amdgcn_wmma_f32_16x16x32_f16(false, a, false, b, (short)0,
                                                      acc[t], false, false);
    }
  }

  // C 16x16 f32 tiles: VGPR r -> row r (lanes 0-15) / r+8 (lanes 16-31), col = lane&15
  int rbase = (rowT << 4) + ((lane >> 4) << 3);
#pragma unroll
  for (int t = 0; t < CT; ++t) {
    int col = ((cg * CT + t) << 4) + (lane & 15);
#pragma unroll
    for (int r = 0; r < 8; ++r)
      C[(size_t)(rbase + r) * Nc + col] = acc[t][r];
  }
}

// ---------------- attention coefficients ----------------
__global__ void k_attn(float* __restrict__ as_, float* __restrict__ ad_,
                       const float* __restrict__ xw,
                       const float* __restrict__ att_s, const float* __restrict__ att_d,
                       int n, int H, int Cc) {
  int i = blockIdx.x * blockDim.x + threadIdx.x;
  if (i >= n * H) return;
  int node = i / H, h = i - node * H;
  const float* xp = xw + (size_t)node * H * Cc + (size_t)h * Cc;
  float s = 0.f, d = 0.f;
  for (int c = 0; c < Cc; ++c) {
    float v = xp[c];
    s += v * att_s[h * Cc + c];
    d += v * att_d[h * Cc + c];
  }
  as_[i] = s; ad_[i] = d;
}

// ---------------- edge softmax + aggregation ----------------

__device__ __forceinline__ void edge_nodes(const long long* __restrict__ ei, int e,
                                           int& s, int& d) {
  if (e < E_EDGES) { s = (int)ei[e]; d = (int)ei[E_EDGES + e]; }
  else             { s = e - E_EDGES; d = s; }   // self loops appended
}

__device__ __forceinline__ float leaky(float v) {
  return v > 0.f ? v : NEG_SLOPE * v;
}

// float atomic max via signed/unsigned int bit tricks (buffer initialized to -inf)
__device__ __forceinline__ void atomicMaxF(float* addr, float val) {
  if (val >= 0.f) atomicMax((int*)addr, __float_as_int(val));
  else            atomicMin((unsigned int*)addr, __float_as_uint(val));
}

__global__ void k_edge_max(float* __restrict__ emax,
                           const float* __restrict__ as_, const float* __restrict__ ad_,
                           const long long* __restrict__ ei, int H) {
  long long i = (long long)blockIdx.x * blockDim.x + threadIdx.x;
  if (i >= (long long)E_TOT * H) return;
  int e = (int)(i / H), h = (int)(i - (long long)e * H);
  int s, d; edge_nodes(ei, e, s, d);
  float ev = leaky(as_[s * H + h] + ad_[d * H + h]);
  atomicMaxF(&emax[d * H + h], ev);
}

__global__ void k_edge_sum(float* __restrict__ denom, const float* __restrict__ emax,
                           const float* __restrict__ as_, const float* __restrict__ ad_,
                           const long long* __restrict__ ei, int H) {
  long long i = (long long)blockIdx.x * blockDim.x + threadIdx.x;
  if (i >= (long long)E_TOT * H) return;
  int e = (int)(i / H), h = (int)(i - (long long)e * H);
  int s, d; edge_nodes(ei, e, s, d);
  float ev = leaky(as_[s * H + h] + ad_[d * H + h]);
  atomicAdd(&denom[d * H + h], __expf(ev - emax[d * H + h]));
}

// one thread = one edge x 4 consecutive channels (Cc % 4 == 0)
__global__ void k_edge_agg(float* __restrict__ out, const float* __restrict__ xw,
                           const float* __restrict__ denom, const float* __restrict__ emax,
                           const float* __restrict__ as_, const float* __restrict__ ad_,
                           const long long* __restrict__ ei, int H, int Cc) {
  int CH  = H * Cc;
  int CH4 = CH >> 2;
  long long i = (long long)blockIdx.x * blockDim.x + threadIdx.x;
  if (i >= (long long)E_TOT * CH4) return;
  int e = (int)(i / CH4), q = (int)(i - (long long)e * CH4);
  int c = q << 2;
  int h = c / Cc;
  int s, d; edge_nodes(ei, e, s, d);
  float ev    = leaky(as_[s * H + h] + ad_[d * H + h]);
  float alpha = __expf(ev - emax[d * H + h]) / (denom[d * H + h] + 1e-16f);
  float4 v = *(const float4*)(xw + (size_t)s * CH + c);
  float* op = out + (size_t)d * CH + c;
  atomicAdd(op + 0, v.x * alpha);
  atomicAdd(op + 1, v.y * alpha);
  atomicAdd(op + 2, v.z * alpha);
  atomicAdd(op + 3, v.w * alpha);
}

// ---------------- ELU(agg + b1) -> f16 ----------------
__global__ void k_elu_bias_f16(_Float16* __restrict__ h16, const float* __restrict__ agg,
                               const float* __restrict__ b, int n, int C) {
  int i = blockIdx.x * blockDim.x + threadIdx.x;
  if (i >= n) return;
  float v = agg[i] + b[i % C];
  v = v > 0.f ? v : (__expf(v) - 1.f);
  h16[i] = (_Float16)v;
}

// ---------------- launcher ----------------

static inline void* carve(char*& p, size_t bytes) {
  void* r = (void*)p;
  p += (bytes + 255) & ~(size_t)255;
  return r;
}

extern "C" void kernel_launch(void* const* d_in, const int* in_sizes, int n_in,
                              void* d_out, int out_size, void* d_ws, size_t ws_size,
                              hipStream_t stream) {
  (void)in_sizes; (void)n_in; (void)out_size; (void)ws_size;
  const float*     x        = (const float*)d_in[0];
  const long long* ei       = (const long long*)d_in[1];   // int64 per reference
  const float*     W1       = (const float*)d_in[2];
  const float*     att_src1 = (const float*)d_in[3];
  const float*     att_dst1 = (const float*)d_in[4];
  const float*     b1       = (const float*)d_in[5];
  const float*     W2       = (const float*)d_in[6];
  const float*     att_src2 = (const float*)d_in[7];
  const float*     att_dst2 = (const float*)d_in[8];
  const float*     b2       = (const float*)d_in[9];
  float*           out      = (float*)d_out;

  char* wp = (char*)d_ws;
  _Float16* x16   = (_Float16*)carve(wp, (size_t)N_NODES * IN_CH * 2);
  _Float16* W1t   = (_Float16*)carve(wp, (size_t)IN_CH * D1 * 2);    // [D1][IN_CH]
  _Float16* W2t   = (_Float16*)carve(wp, (size_t)D1 * OUT_CH * 2);   // [OUT_CH][D1]
  float*    xw1   = (float*)carve(wp, (size_t)N_NODES * D1 * 4);
  float*    agg1  = (float*)carve(wp, (size_t)N_NODES * D1 * 4);
  _Float16* h16   = (_Float16*)carve(wp, (size_t)N_NODES * D1 * 2);
  float*    xw2   = (float*)carve(wp, (size_t)N_NODES * OUT_CH * 4);
  float*    as1   = (float*)carve(wp, (size_t)N_NODES * HEADS1 * 4);
  float*    ad1   = (float*)carve(wp, (size_t)N_NODES * HEADS1 * 4);
  float*    emax1 = (float*)carve(wp, (size_t)N_NODES * HEADS1 * 4);
  float*    den1  = (float*)carve(wp, (size_t)N_NODES * HEADS1 * 4);
  float*    as2   = (float*)carve(wp, (size_t)N_NODES * 4);
  float*    ad2   = (float*)carve(wp, (size_t)N_NODES * 4);
  float*    emax2 = (float*)carve(wp, (size_t)N_NODES * 4);
  float*    den2  = (float*)carve(wp, (size_t)N_NODES * 4);

  const int B = 256;
  auto g = [](long long n, int b) { return (unsigned)((n + b - 1) / b); };

  // ---- layer 1 ----
  k_f32_to_f16<<<g((long long)N_NODES * IN_CH, B), B, 0, stream>>>(x16, x, N_NODES * IN_CH);
  k_f32_to_f16_T<<<g(IN_CH * D1, B), B, 0, stream>>>(W1t, W1, IN_CH, D1);

  {
    int tiles = (N_NODES / 16) * (D1 / 64);          // CT=4 -> 6250 wave-tiles
    k_gemm_wmma<IN_CH, 4><<<g(tiles, B / 32), B, 0, stream>>>(xw1, x16, W1t, N_NODES, D1);
  }

  k_attn<<<g((long long)N_NODES * HEADS1, B), B, 0, stream>>>(
      as1, ad1, xw1, att_src1, att_dst1, N_NODES, HEADS1, HID1);

  k_fill<<<g((long long)N_NODES * HEADS1, B), B, 0, stream>>>(emax1, -INFINITY, N_NODES * HEADS1);
  k_fill<<<g((long long)N_NODES * HEADS1, B), B, 0, stream>>>(den1, 0.f, N_NODES * HEADS1);
  k_fill<<<g((long long)N_NODES * D1, B), B, 0, stream>>>(agg1, 0.f, N_NODES * D1);

  k_edge_max<<<g((long long)E_TOT * HEADS1, B), B, 0, stream>>>(emax1, as1, ad1, ei, HEADS1);
  k_edge_sum<<<g((long long)E_TOT * HEADS1, B), B, 0, stream>>>(den1, emax1, as1, ad1, ei, HEADS1);
  k_edge_agg<<<g((long long)E_TOT * (D1 / 4), B), B, 0, stream>>>(
      agg1, xw1, den1, emax1, as1, ad1, ei, HEADS1, HID1);

  k_elu_bias_f16<<<g((long long)N_NODES * D1, B), B, 0, stream>>>(h16, agg1, b1, N_NODES * D1, D1);

  // ---- layer 2 (heads=1, mean over 1 head == identity) ----
  k_f32_to_f16_T<<<g(D1 * OUT_CH, B), B, 0, stream>>>(W2t, W2, D1, OUT_CH);

  {
    int tiles = (N_NODES / 16) * (OUT_CH / 32);      // CT=2 -> 3125 wave-tiles
    k_gemm_wmma<D1, 2><<<g(tiles, B / 32), B, 0, stream>>>(xw2, h16, W2t, N_NODES, OUT_CH);
  }

  k_attn<<<g(N_NODES, B), B, 0, stream>>>(as2, ad2, xw2, att_src2, att_dst2, N_NODES, 1, OUT_CH);

  k_fill<<<g(N_NODES, B), B, 0, stream>>>(emax2, -INFINITY, N_NODES);
  k_fill<<<g(N_NODES, B), B, 0, stream>>>(den2, 0.f, N_NODES);
  k_bias_fill<<<g((long long)N_NODES * OUT_CH, B), B, 0, stream>>>(out, b2, N_NODES * OUT_CH, OUT_CH);

  k_edge_max<<<g(E_TOT, B), B, 0, stream>>>(emax2, as2, ad2, ei, 1);
  k_edge_sum<<<g(E_TOT, B), B, 0, stream>>>(den2, emax2, as2, ad2, ei, 1);
  k_edge_agg<<<g((long long)E_TOT * (OUT_CH / 4), B), B, 0, stream>>>(
      out, xw2, den2, emax2, as2, ad2, ei, 1, OUT_CH);
}
